// Decoder_19000935317880
// MI455X (gfx1250) — matrix operand order repaired and checked by reading, not compiled, not run on hardware
//
#include <hip/hip_runtime.h>

// ---------------------------------------------------------------------------
// CDNA5 (gfx1250) transformer decoder block.
// All matmuls -> v_wmma_f32_16x16x32_bf16 (wave32 WMMA), all operands staged
// so that every LDS fragment read is a ds_load_b128. Global->LDS staging uses
// the gfx1250 async-to-LDS path (ASYNCcnt) when the toolchain exposes it.
// ---------------------------------------------------------------------------

typedef __attribute__((ext_vector_type(16))) __bf16 v16bf;
typedef __attribute__((ext_vector_type(8)))  float  v8f;
typedef __attribute__((ext_vector_type(4)))  int    v4i;
typedef unsigned short u16t;
typedef unsigned int   u32t;
typedef unsigned long long u64t;

// pointer-to-addrspace typedefs for the async-to-LDS builtin
typedef __attribute__((address_space(1))) v4i* gv4ip;   // global int4*
typedef __attribute__((address_space(3))) v4i* lv4ip;   // LDS int4*

struct alignas(16) U128 { u32t w[4]; };

#if defined(__has_builtin)
#  if __has_builtin(__builtin_amdgcn_global_load_async_to_lds_b128)
#    define HAVE_ASYNC_LDS 1
#  endif
#endif
#ifndef HAVE_ASYNC_LDS
#  define HAVE_ASYNC_LDS 0
#endif

__device__ __forceinline__ u16t f2bf(float f) {
    u32t u = __builtin_bit_cast(u32t, f);
    u32t r = u + 0x7FFFu + ((u >> 16) & 1u);   // round-to-nearest-even
    return (u16t)(r >> 16);
}

// 16-byte global -> LDS copy; async (ASYNCcnt) when available.
__device__ __forceinline__ void copy16_g2l(const u16t* g, u16t* l) {
#if HAVE_ASYNC_LDS
    __builtin_amdgcn_global_load_async_to_lds_b128(
        (gv4ip)(u64t)(uintptr_t)g,
        (lv4ip)(u32t)(uintptr_t)l,
        0, 0);
#else
    *reinterpret_cast<U128*>(l) = *reinterpret_cast<const U128*>(g);
#endif
}

__device__ __forceinline__ void wait_async_all() {
#if HAVE_ASYNC_LDS
#  if defined(__has_builtin) && __has_builtin(__builtin_amdgcn_s_wait_asynccnt)
    __builtin_amdgcn_s_wait_asynccnt(0);
#  else
    asm volatile("s_wait_asynccnt 0" ::: "memory");
#  endif
#endif
}

// ---------------------------------------------------------------------------
// fp32 -> bf16 (activations)
// ---------------------------------------------------------------------------
__global__ void cvt_f32_bf16(const float* __restrict__ in, u16t* __restrict__ out, long n) {
    long i = (long)blockIdx.x * blockDim.x + threadIdx.x;
    if (i < n) out[i] = f2bf(in[i]);
}

// ---------------------------------------------------------------------------
// fused fp32 -> bf16 transpose: out[n*K + k] = bf16(in[k*N + n]); 32x32 tiles
// ---------------------------------------------------------------------------
__global__ void __launch_bounds__(256)
cvt_transpose(const float* __restrict__ in, u16t* __restrict__ out, int K, int N) {
    __shared__ float tile[32][33];
    int bn = blockIdx.x * 32, bk = blockIdx.y * 32;
    int tx = threadIdx.x & 31, ty = threadIdx.x >> 5;      // 32 x 8
    #pragma unroll
    for (int i = 0; i < 4; ++i)
        tile[ty + i * 8][tx] = in[(size_t)(bk + ty + i * 8) * N + bn + tx];
    __syncthreads();
    #pragma unroll
    for (int i = 0; i < 4; ++i)
        out[(size_t)(bn + ty + i * 8) * K + bk + tx] = f2bf(tile[tx][ty + i * 8]);
}

// ---------------------------------------------------------------------------
// Tiled WMMA GEMM:  C[M,N] = A[M,K] @ B[N,K]^T (+ bias[n]) (ReLU?)
// B is ALWAYS stored [N,K] ("BT" form) so A and B fragment loads are b128.
// Block tile 128x128xBK32, 256 threads = 8 waves (4Mx2N), wave tile 32x64 =
// 2x4 WMMA tiles = 8 v_wmma per K-step. Double-buffered LDS + async staging.
// OUT_TRANS stores C transposed ([N,M]) with contiguous 16B stores.
// ---------------------------------------------------------------------------
template<bool OUT_BF16, bool RELU, bool OUT_TRANS>
__global__ void __launch_bounds__(256)
gemm_wmma(const u16t* __restrict__ A, int lda, long sA,
          const u16t* __restrict__ Bm, int ldb, long sB,
          void* __restrict__ Cp, int ldc, long sC,
          const float* __restrict__ bias, int K)
{
    constexpr int BM = 128, BN = 128, BK = 32, PAD = 8, LDK = BK + PAD;
    __shared__ alignas(16) u16t As[2][BM][LDK];   // [m][k]
    __shared__ alignas(16) u16t Bs[2][BN][LDK];   // [n][k]

    const int tid  = threadIdx.x;
    const int lane = tid & 31;
    const int wid  = tid >> 5;
    const int wm   = wid >> 1;          // 0..3  -> m origin wm*32
    const int wn   = wid & 1;           // 0..1  -> n origin wn*64
    const int half = lane >> 4;         // K-half select (ISA 7.12.2)
    const int lr   = lane & 15;

    const int blockN = blockIdx.x * BN;
    const int blockM = blockIdx.y * BM;
    const int batch  = blockIdx.z;
    A  += (long)batch * sA;
    Bm += (long)batch * sB;

    const int row = tid >> 1;           // 0..127
    const int col = (tid & 1) * 16;     // 0 or 16

    auto stage = [&](int kt, int buf) {
        const u16t* a = A + (size_t)(blockM + row) * lda + kt * BK + col;
        copy16_g2l(a,     &As[buf][row][col]);
        copy16_g2l(a + 8, &As[buf][row][col + 8]);
        const u16t* b = Bm + (size_t)(blockN + row) * ldb + kt * BK + col;
        copy16_g2l(b,     &Bs[buf][row][col]);
        copy16_g2l(b + 8, &Bs[buf][row][col + 8]);
    };

    v8f acc[2][4] = {};
    const int NK = K / BK;

    stage(0, 0);
    for (int kt = 0; kt < NK; ++kt) {
        const int cur = kt & 1;
        wait_async_all();
        __syncthreads();
        if (kt + 1 < NK) stage(kt + 1, cur ^ 1);

        v16bf af[2], bf[4];
        #pragma unroll
        for (int tm = 0; tm < 2; ++tm) {
            int m = wm * 32 + tm * 16 + lr;
            union { U128 q[2]; v16bf v; } u;
            u.q[0] = *reinterpret_cast<const U128*>(&As[cur][m][half * 8]);       // K=8h+0..7
            u.q[1] = *reinterpret_cast<const U128*>(&As[cur][m][16 + half * 8]);  // K=16+8h+0..7
            af[tm] = u.v;
        }
        #pragma unroll
        for (int tn = 0; tn < 4; ++tn) {
            int n = wn * 64 + tn * 16 + lr;
            union { U128 q[2]; v16bf v; } u;
            u.q[0] = *reinterpret_cast<const U128*>(&Bs[cur][n][half * 8]);       // K=8h+0..7
            u.q[1] = *reinterpret_cast<const U128*>(&Bs[cur][n][16 + half * 8]);  // K=16+8h+0..7
            bf[tn] = u.v;
        }
        #pragma unroll
        for (int tm = 0; tm < 2; ++tm)
            #pragma unroll
            for (int tn = 0; tn < 4; ++tn)
                acc[tm][tn] = __builtin_amdgcn_wmma_f32_16x16x32_bf16(
                    false, af[tm], false, bf[tn],
                    (short)0, acc[tm][tn], false, false);
        __syncthreads();
    }

    // ---- epilogue (C frag layout: m = 8*half + r, n = lr within a tile) ----
    #pragma unroll
    for (int tm = 0; tm < 2; ++tm) {
        #pragma unroll
        for (int tn = 0; tn < 4; ++tn) {
            int n = blockN + wn * 64 + tn * 16 + lr;
            float bv = bias ? bias[n] : 0.0f;
            if (OUT_TRANS) {
                // 8 consecutive m per lane -> one 16B store at C[n][m0..m0+7]
                int m0 = blockM + wm * 32 + tm * 16 + half * 8;
                union { U128 q; u16t s[8]; } u;
                #pragma unroll
                for (int r = 0; r < 8; ++r) {
                    float v = acc[tm][tn][r] + bv;
                    if (RELU) v = fmaxf(v, 0.0f);
                    u.s[r] = f2bf(v);
                }
                *reinterpret_cast<U128*>(
                    reinterpret_cast<u16t*>(Cp) + (size_t)batch * sC + (size_t)n * ldc + m0) = u.q;
            } else {
                #pragma unroll
                for (int r = 0; r < 8; ++r) {
                    int m = blockM + wm * 32 + tm * 16 + half * 8 + r;
                    float v = acc[tm][tn][r] + bv;
                    if (RELU) v = fmaxf(v, 0.0f);
                    size_t idx = (size_t)batch * sC + (size_t)m * ldc + n;
                    if (OUT_BF16) reinterpret_cast<u16t*>(Cp)[idx] = f2bf(v);
                    else          reinterpret_cast<float*>(Cp)[idx] = v;
                }
            }
        }
    }
}

// ---------------------------------------------------------------------------
// Softmax over 512-wide rows; causal mask matches reference (-1e6 fill).
// ---------------------------------------------------------------------------
template<bool CAUSAL>
__global__ void __launch_bounds__(256)
softmax_rows(const float* __restrict__ S, u16t* __restrict__ P, float scale)
{
    __shared__ float red[256];
    const int NC = 512;
    int row = blockIdx.x;
    int q   = row & (NC - 1);
    const float* s = S + (size_t)row * NC;
    u16t*        p = P + (size_t)row * NC;
    int t = threadIdx.x;
    int c0 = t, c1 = t + 256;
    float a0 = s[c0] * scale, a1 = s[c1] * scale;
    if (CAUSAL) {
        if (c0 > q) a0 = -1.0e6f;
        if (c1 > q) a1 = -1.0e6f;
    }
    red[t] = fmaxf(a0, a1); __syncthreads();
    for (int st = 128; st > 0; st >>= 1) { if (t < st) red[t] = fmaxf(red[t], red[t + st]); __syncthreads(); }
    float m = red[0]; __syncthreads();
    float e0 = __expf(a0 - m), e1 = __expf(a1 - m);
    red[t] = e0 + e1; __syncthreads();
    for (int st = 128; st > 0; st >>= 1) { if (t < st) red[t] += red[t + st]; __syncthreads(); }
    float inv = 1.0f / red[0];
    p[c0] = f2bf(e0 * inv);
    p[c1] = f2bf(e1 * inv);
}

// ---------------------------------------------------------------------------
// out = LayerNorm(y + res) * g + b ; fp32 out (+ optional bf16 copy)
// ---------------------------------------------------------------------------
__global__ void __launch_bounds__(256)
add_ln(const float* __restrict__ y, const float* __restrict__ res,
       const float* __restrict__ g, const float* __restrict__ be,
       float* __restrict__ outf, u16t* __restrict__ outb)
{
    const int D = 512;
    __shared__ float red[256];
    int row = blockIdx.x, t = threadIdx.x;
    size_t base = (size_t)row * D;
    int c0 = t, c1 = t + 256;
    float x0 = y[base + c0] + res[base + c0];
    float x1 = y[base + c1] + res[base + c1];
    red[t] = x0 + x1; __syncthreads();
    for (int st = 128; st > 0; st >>= 1) { if (t < st) red[t] += red[t + st]; __syncthreads(); }
    float mu = red[0] * (1.0f / D); __syncthreads();
    float d0 = x0 - mu, d1 = x1 - mu;
    red[t] = d0 * d0 + d1 * d1; __syncthreads();
    for (int st = 128; st > 0; st >>= 1) { if (t < st) red[t] += red[t + st]; __syncthreads(); }
    float rstd = rsqrtf(red[0] * (1.0f / D) + 1e-5f);
    float o0 = g[c0] * d0 * rstd + be[c0];
    float o1 = g[c1] * d1 * rstd + be[c1];
    outf[base + c0] = o0;
    outf[base + c1] = o1;
    if (outb) { outb[base + c0] = f2bf(o0); outb[base + c1] = f2bf(o1); }
}

// ---------------------------------------------------------------------------
extern "C" void kernel_launch(void* const* d_in, const int* in_sizes, int n_in,
                              void* d_out, int out_size, void* d_ws, size_t ws_size,
                              hipStream_t stream)
{
    (void)in_sizes; (void)n_in; (void)out_size; (void)ws_size;
    constexpr int  B  = 8, S = 512, D = 512, HD = 4096;
    constexpr long BS = (long)B * S;                    // 4096 flattened rows
    const float SCALE = 0.04419417382415922f;           // 1/sqrt(512)

    const float* dec = (const float*)d_in[0];
    const float* enc = (const float*)d_in[1];
    const float* Wq  = (const float*)d_in[3];  const float* bq  = (const float*)d_in[4];
    const float* Wk  = (const float*)d_in[5];  const float* bk  = (const float*)d_in[6];
    const float* Wv  = (const float*)d_in[7];  const float* bv  = (const float*)d_in[8];
    const float* Wo0 = (const float*)d_in[9];  const float* bo0 = (const float*)d_in[10];
    const float* g0  = (const float*)d_in[11]; const float* be0 = (const float*)d_in[12];
    const float* Wq2 = (const float*)d_in[13]; const float* bq2 = (const float*)d_in[14];
    const float* Wk2 = (const float*)d_in[15]; const float* bk2 = (const float*)d_in[16];
    const float* Wv2 = (const float*)d_in[17]; const float* bv2 = (const float*)d_in[18];
    const float* Wo1 = (const float*)d_in[19]; const float* bo1 = (const float*)d_in[20];
    const float* g1  = (const float*)d_in[21]; const float* be1 = (const float*)d_in[22];
    const float* Wf0 = (const float*)d_in[23]; const float* bf0 = (const float*)d_in[24];
    const float* Wf1 = (const float*)d_in[25]; const float* bf1 = (const float*)d_in[26];
    const float* g2  = (const float*)d_in[27]; const float* be2 = (const float*)d_in[28];

    // -------- workspace layout --------
    char* wsp = (char*)d_ws;
    size_t off = 0;
    auto alloc = [&](size_t bytes) -> void* {
        void* p = wsp + off;
        off += (bytes + 255) & ~(size_t)255;
        return p;
    };
    // weights, pre-transposed to [N,K] bf16
    u16t* wqT  = (u16t*)alloc((size_t)D * HD * 2);
    u16t* wkT  = (u16t*)alloc((size_t)D * HD * 2);
    u16t* wvT  = (u16t*)alloc((size_t)D * HD * 2);
    u16t* wo0T = (u16t*)alloc((size_t)HD * D * 2);
    u16t* wq2T = (u16t*)alloc((size_t)D * HD * 2);
    u16t* wk2T = (u16t*)alloc((size_t)D * HD * 2);
    u16t* wv2T = (u16t*)alloc((size_t)D * HD * 2);
    u16t* wo1T = (u16t*)alloc((size_t)HD * D * 2);
    u16t* wf0T = (u16t*)alloc((size_t)D * D * 2);
    u16t* wf1T = (u16t*)alloc((size_t)D * D * 2);
    u16t* dec_bf = (u16t*)alloc((size_t)BS * D * 2);
    u16t* enc_bf = (u16t*)alloc((size_t)BS * D * 2);
    u16t* q_bf   = (u16t*)alloc((size_t)BS * HD * 2);   // reused by layer 2
    u16t* k_bf   = (u16t*)alloc((size_t)BS * HD * 2);
    u16t* vt_bf  = (u16t*)alloc((size_t)BS * HD * 2);   // V^T: [HD][B*S]
    float* scores = (float*)alloc((size_t)B * S * S * 4);
    u16t* probs   = (u16t*)alloc((size_t)B * S * S * 2);
    u16t* ctx_bf  = (u16t*)alloc((size_t)BS * HD * 2);
    float* proj   = (float*)alloc((size_t)BS * D * 4);
    float* x0f    = (float*)alloc((size_t)BS * D * 4);
    u16t* x0b     = (u16t*)alloc((size_t)BS * D * 2);
    float* x1f    = (float*)alloc((size_t)BS * D * 4);
    u16t* x1b     = (u16t*)alloc((size_t)BS * D * 2);
    u16t* hb      = (u16t*)alloc((size_t)BS * D * 2);

    dim3 blk(256);
    auto cvt = [&](const float* src, u16t* dst, long n) {
        cvt_f32_bf16<<<dim3((unsigned)((n + 255) / 256)), blk, 0, stream>>>(src, dst, n);
    };
    auto cvtT = [&](const float* src, u16t* dst, int K, int N) {
        cvt_transpose<<<dim3(N / 32, K / 32), blk, 0, stream>>>(src, dst, K, N);
    };

    // -------- conversions / weight transposes --------
    cvt(dec, dec_bf, BS * D);
    cvt(enc, enc_bf, BS * D);
    cvtT(Wq,  wqT,  D, HD);  cvtT(Wk,  wkT,  D, HD);  cvtT(Wv,  wvT,  D, HD);
    cvtT(Wo0, wo0T, HD, D);
    cvtT(Wq2, wq2T, D, HD);  cvtT(Wk2, wk2T, D, HD);  cvtT(Wv2, wv2T, D, HD);
    cvtT(Wo1, wo1T, HD, D);
    cvtT(Wf0, wf0T, D, D);   cvtT(Wf1, wf1T, D, D);

    // ================= layer 1: masked self-attention =================
    gemm_wmma<true, false, false><<<dim3(HD / 128, BS / 128, 1), blk, 0, stream>>>(
        dec_bf, D, 0, wqT, D, 0, q_bf, HD, 0, bq, D);
    gemm_wmma<true, false, false><<<dim3(HD / 128, BS / 128, 1), blk, 0, stream>>>(
        dec_bf, D, 0, wkT, D, 0, k_bf, HD, 0, bk, D);
    gemm_wmma<true, false, true><<<dim3(HD / 128, BS / 128, 1), blk, 0, stream>>>(
        dec_bf, D, 0, wvT, D, 0, vt_bf, (int)BS, 0, bv, D);       // V^T [HD][BS]

    gemm_wmma<false, false, false><<<dim3(S / 128, S / 128, B), blk, 0, stream>>>(
        q_bf, HD, (long)S * HD, k_bf, HD, (long)S * HD,
        scores, S, (long)S * S, nullptr, HD);
    softmax_rows<true><<<dim3(B * S), blk, 0, stream>>>(scores, probs, SCALE);

    gemm_wmma<true, false, false><<<dim3(HD / 128, S / 128, B), blk, 0, stream>>>(
        probs, S, (long)S * S, vt_bf, (int)BS, (long)S,           // batch col offset
        ctx_bf, HD, (long)S * HD, nullptr, S);

    gemm_wmma<false, false, false><<<dim3(D / 128, BS / 128, 1), blk, 0, stream>>>(
        ctx_bf, HD, 0, wo0T, HD, 0, proj, D, 0, bo0, HD);
    add_ln<<<dim3(BS), blk, 0, stream>>>(proj, dec, g0, be0, x0f, x0b);

    // ================= layer 2: cross-attention =================
    gemm_wmma<true, false, false><<<dim3(HD / 128, BS / 128, 1), blk, 0, stream>>>(
        x0b, D, 0, wq2T, D, 0, q_bf, HD, 0, bq2, D);
    gemm_wmma<true, false, false><<<dim3(HD / 128, BS / 128, 1), blk, 0, stream>>>(
        enc_bf, D, 0, wk2T, D, 0, k_bf, HD, 0, bk2, D);
    gemm_wmma<true, false, true><<<dim3(HD / 128, BS / 128, 1), blk, 0, stream>>>(
        enc_bf, D, 0, wv2T, D, 0, vt_bf, (int)BS, 0, bv2, D);

    gemm_wmma<false, false, false><<<dim3(S / 128, S / 128, B), blk, 0, stream>>>(
        q_bf, HD, (long)S * HD, k_bf, HD, (long)S * HD,
        scores, S, (long)S * S, nullptr, HD);
    softmax_rows<false><<<dim3(B * S), blk, 0, stream>>>(scores, probs, SCALE);

    gemm_wmma<true, false, false><<<dim3(HD / 128, S / 128, B), blk, 0, stream>>>(
        probs, S, (long)S * S, vt_bf, (int)BS, (long)S,
        ctx_bf, HD, (long)S * HD, nullptr, S);

    gemm_wmma<false, false, false><<<dim3(D / 128, BS / 128, 1), blk, 0, stream>>>(
        ctx_bf, HD, 0, wo1T, HD, 0, proj, D, 0, bo1, HD);
    add_ln<<<dim3(BS), blk, 0, stream>>>(proj, x0f, g1, be1, x1f, x1b);

    // ================= FFN =================
    gemm_wmma<true, true, false><<<dim3(D / 128, BS / 128, 1), blk, 0, stream>>>(
        x1b, D, 0, wf0T, D, 0, hb, D, 0, bf0, D);
    gemm_wmma<false, false, false><<<dim3(D / 128, BS / 128, 1), blk, 0, stream>>>(
        hb, D, 0, wf1T, D, 0, proj, D, 0, bf1, D);
    add_ln<<<dim3(BS), blk, 0, stream>>>(proj, x1f, g2, be2, (float*)d_out, nullptr);
}
